// RAE_51908974739480
// MI455X (gfx1250) — matrix operand order repaired and checked by reading, not compiled, not run on hardware
//
#include <hip/hip_runtime.h>
#include <hip/hip_bf16.h>

typedef __bf16 bf16_t;
typedef __attribute__((ext_vector_type(16))) __bf16 v16bf;
typedef __attribute__((ext_vector_type(8)))  __bf16 v8bf;
typedef __attribute__((ext_vector_type(8)))  float  v8f;

__device__ __forceinline__ bf16_t f2bf(float f) {
    unsigned u = __float_as_uint(f);
    unsigned r = u + 0x7FFFu + ((u >> 16) & 1u);
    unsigned short h = (unsigned short)(r >> 16);
    bf16_t b; __builtin_memcpy(&b, &h, 2); return b;
}
__device__ __forceinline__ float bf2f(bf16_t b) {
    unsigned short h; __builtin_memcpy(&h, &b, 2);
    return __uint_as_float(((unsigned)h) << 16);
}
__device__ __forceinline__ v8f wmma_bf16(v16bf a, v16bf b, v8f c) {
    return __builtin_amdgcn_wmma_f32_16x16x32_bf16(false, a, false, b, (short)0, c, false, false);
}
__device__ __forceinline__ v16bf cat16(v8bf a, v8bf b) {
    return __builtin_shufflevector(a, b, 0,1,2,3,4,5,6,7,8,9,10,11,12,13,14,15);
}

// ---------------- conv1: (B,3,64,64) f32 NCHW -> (B,32,32,128) bf16 NHWC, relu ----
__global__ void k_conv1(const float* __restrict__ x, const float* __restrict__ W,
                        const float* __restrict__ b, bf16_t* __restrict__ out) {
    int idx = blockIdx.x * 256 + threadIdx.x;          // B*32*32*128
    int co = idx & 127;
    int pix = idx >> 7;
    int wo = pix & 31, ho = (pix >> 5) & 31, n = pix >> 10;
    float acc = b[co];
    for (int ci = 0; ci < 3; ++ci)
        for (int kh = 0; kh < 3; ++kh) {
            int ih = 2 * ho - 1 + kh;
            if ((unsigned)ih >= 64u) continue;
            for (int kw = 0; kw < 3; ++kw) {
                int iw = 2 * wo - 1 + kw;
                if ((unsigned)iw >= 64u) continue;
                acc += x[((n * 3 + ci) * 64 + ih) * 64 + iw] *
                       W[((co * 3 + ci) * 3 + kh) * 3 + kw];
            }
        }
    out[((size_t)pix << 7) | co] = f2bf(fmaxf(acc, 0.f));
}

// ------------- pack conv/deconv weights (O,I,3,3) f32 -> B-fragments bf16 --------
// frag[((r*kcN + kc)*tilesN + nt)*512 + lane*16 + e], K=ci, N=co
__global__ void k_pack_w(const float* __restrict__ W, bf16_t* __restrict__ frag,
                         int lgCin, int lgCout, int total) {
    int idx = blockIdx.x * 256 + threadIdx.x;
    if (idx >= total) return;
    int lgTn = lgCout - 4, lgKc = lgCin - 5;
    int e    = idx & 15;
    int lane = (idx >> 4) & 31;
    int nt   = (idx >> 9) & ((1 << lgTn) - 1);
    int kc   = (idx >> (9 + lgTn)) & ((1 << lgKc) - 1);
    int r    = idx >> (9 + lgTn + lgKc);
    int Cin  = 1 << lgCin;
    int ci = (kc << 5) + ((e >> 3) << 4) + ((lane >> 4) << 3) + (e & 7);
    int co = (nt << 4) + (lane & 15);
    frag[idx] = f2bf(W[(co * Cin + ci) * 9 + r]);
}

// ---- pack prototypes (G=4,P=64,D=4096) for dists GEMM: K=d, N=g*64+p -----------
__global__ void k_pack_pd(const float* __restrict__ P, bf16_t* __restrict__ frag) {
    int idx = blockIdx.x * 256 + threadIdx.x;          // 128*16*512 = 1048576
    int e = idx & 15, lane = (idx >> 4) & 31;
    int nt = (idx >> 9) & 15;
    int kc = idx >> 13;
    int d  = (kc << 5) + ((e >> 3) << 4) + ((lane >> 4) << 3) + (e & 7);
    int gp = (nt << 4) + (lane & 15);
    frag[idx] = f2bf(P[(size_t)gp * 4096 + d]);
}

// ---- pack prototypes for mixing GEMM: per g, K=p, N=d --------------------------
__global__ void k_pack_pm(const float* __restrict__ P, bf16_t* __restrict__ frag) {
    int idx = blockIdx.x * 256 + threadIdx.x;          // 4*2*256*512 = 1048576
    int e = idx & 15, lane = (idx >> 4) & 31;
    int dt = (idx >> 9) & 255;
    int kc = (idx >> 17) & 1;
    int g  = idx >> 18;
    int p = (kc << 5) + ((e >> 3) << 4) + ((lane >> 4) << 3) + (e & 7);
    int d = (dt << 4) + (lane & 15);
    frag[idx] = f2bf(P[((size_t)g * 64 + p) * 4096 + d]);
}

// ---- squared row norms of a (rows x K) f32 matrix ------------------------------
__global__ void k_rownorm(const float* __restrict__ M, float* __restrict__ out, int K) {
    int r = blockIdx.x * blockDim.x + threadIdx.x;
    const float* p = M + (size_t)r * K;
    float s = 0.f;
    for (int i = 0; i < K; ++i) { float v = p[i]; s += v * v; }
    out[r] = s;
}

// ============== implicit-GEMM conv (MODE=0: stride2 pad1) / deconv (MODE=1) =====
// in: NHWC bf16 (B,Hi,Hi,Cin); out: NHWC bf16 (B,Ho,Ho,Cout)
// 2x2 register tiling: each wave computes a 32(M) x 32(N) output tile.
template<int MODE, int ACT, bool EXTRA>
__global__ __launch_bounds__(256)
void k_convgemm(const bf16_t* __restrict__ in, const bf16_t* __restrict__ wfrag,
                const float* __restrict__ bias, bf16_t* __restrict__ out_nhwc,
                float* __restrict__ out_nchw_f32, bf16_t* __restrict__ out_nchw_bf,
                int lgHi, int lgHo, int lgCin, int lgCout) {
    const int Hi = 1 << lgHi, Ho = 1 << lgHo;
    const int tilesN  = 1 << (lgCout - 4);   // 16-col tiles
    const int tilesN2 = 1 << (lgCout - 5);   // 32-col super tiles
    const int kcN = 1 << (lgCin - 5);
    const int lane = threadIdx.x & 31;
    const int wtile = (blockIdx.x << 3) + (threadIdx.x >> 5);
    const int nt2 = wtile & (tilesN2 - 1);
    const int mt2 = wtile >> (lgCout - 5);
    const int col = lane & 15;
    const int khalf = lane >> 4;
    const int klo = khalf << 3;

    // two A rows handled by this lane
    const int m0 = (mt2 << 5) + col;
    const int m1 = m0 + 16;
    const int wo0 = m0 & (Ho - 1), ho0 = (m0 >> lgHo) & (Ho - 1), n0 = m0 >> (2 * lgHo);
    const int wo1 = m1 & (Ho - 1), ho1 = (m1 >> lgHo) & (Ho - 1), n1 = m1 >> (2 * lgHo);

    v8f c[2][2];
    #pragma unroll
    for (int i = 0; i < 2; ++i)
        #pragma unroll
        for (int j = 0; j < 2; ++j)
            #pragma unroll
            for (int k = 0; k < 8; ++k) c[i][j][k] = 0.f;
    v8bf zz;
    #pragma unroll
    for (int i = 0; i < 8; ++i) zz[i] = f2bf(0.f);

    #pragma unroll
    for (int kh = 0; kh < 3; ++kh) {
        #pragma unroll
        for (int kw = 0; kw < 3; ++kw) {
            int r = kh * 3 + kw;
            int ih0, iw0, ih1, iw1; bool v0, v1;
            if (MODE == 0) {
                ih0 = 2 * ho0 - 1 + kh; iw0 = 2 * wo0 - 1 + kw;
                ih1 = 2 * ho1 - 1 + kh; iw1 = 2 * wo1 - 1 + kw;
                v0 = ((unsigned)ih0 < (unsigned)Hi) && ((unsigned)iw0 < (unsigned)Hi);
                v1 = ((unsigned)ih1 < (unsigned)Hi) && ((unsigned)iw1 < (unsigned)Hi);
            } else {
                int th0 = ho0 - 2 + kh, tw0 = wo0 - 2 + kw;
                int th1 = ho1 - 2 + kh, tw1 = wo1 - 2 + kw;
                ih0 = th0 >> 1; iw0 = tw0 >> 1;
                ih1 = th1 >> 1; iw1 = tw1 >> 1;
                v0 = !(th0 & 1) && !(tw0 & 1) &&
                     ((unsigned)ih0 < (unsigned)Hi) && ((unsigned)iw0 < (unsigned)Hi);
                v1 = !(th1 & 1) && !(tw1 & 1) &&
                     ((unsigned)ih1 < (unsigned)Hi) && ((unsigned)iw1 < (unsigned)Hi);
            }
            const bf16_t* ab0 =
                in + ((((size_t)((n0 << lgHi) + ih0) << lgHi) + iw0) << lgCin) + klo;
            const bf16_t* ab1 =
                in + ((((size_t)((n1 << lgHi) + ih1) << lgHi) + iw1) << lgCin) + klo;
            const bf16_t* wb =
                wfrag + (((size_t)(r * kcN) * tilesN + (nt2 << 1)) * 32 + lane) * 16;
            for (int kc = 0; kc < kcN; ++kc) {
                v8bf a00, a01, a10, a11;
                if (v0) { a00 = *(const v8bf*)(ab0 + (kc << 5));
                          a01 = *(const v8bf*)(ab0 + (kc << 5) + 16); }
                else    { a00 = zz; a01 = zz; }
                if (v1) { a10 = *(const v8bf*)(ab1 + (kc << 5));
                          a11 = *(const v8bf*)(ab1 + (kc << 5) + 16); }
                else    { a10 = zz; a11 = zz; }
                v16bf A0 = cat16(a00, a01);
                v16bf A1 = cat16(a10, a11);
                const bf16_t* wkc = wb + ((size_t)kc * tilesN << 9);
                v16bf B0 = *(const v16bf*)(wkc);
                v16bf B1 = *(const v16bf*)(wkc + 512);
                c[0][0] = wmma_bf16(A0, B0, c[0][0]);
                c[0][1] = wmma_bf16(A0, B1, c[0][1]);
                c[1][0] = wmma_bf16(A1, B0, c[1][0]);
                c[1][1] = wmma_bf16(A1, B1, c[1][1]);
            }
        }
    }
    // epilogue: bias + activation, write NHWC (and optional NCHW f32/bf16)
    #pragma unroll
    for (int tj = 0; tj < 2; ++tj) {
        const int co = (nt2 << 5) + (tj << 4) + col;
        const float bv = bias[co];
        #pragma unroll
        for (int ti = 0; ti < 2; ++ti) {
            const int rowbase = (mt2 << 5) + (ti << 4) + (khalf << 3);
            #pragma unroll
            for (int v = 0; v < 8; ++v) {
                int row = rowbase + v;
                int wo2 = row & (Ho - 1);
                int ho2 = (row >> lgHo) & (Ho - 1);
                int n2  = row >> (2 * lgHo);
                float f = c[ti][tj][v] + bv;
                if (ACT == 1) f = fmaxf(f, 0.f);
                else if (ACT == 2) f = 1.f / (1.f + __expf(-f));
                size_t onhwc = ((((size_t)((n2 << lgHo) + ho2) << lgHo) + wo2) << lgCout) + co;
                out_nhwc[onhwc] = f2bf(f);
                if (EXTRA) {
                    size_t onchw = ((((size_t)((n2 << lgCout) + co) << lgHo) + ho2) << lgHo) + wo2;
                    out_nchw_f32[onchw] = f;
                    out_nchw_bf[onchw]  = f2bf(f);
                }
            }
        }
    }
}

// ============== dists GEMM: (256x4096)x(4096x256) + norms epilogue ==============
// 2x2 register tiling: 32x32 per wave, 64 super-tiles total.
__global__ __launch_bounds__(256)
void k_dists(const bf16_t* __restrict__ z, const bf16_t* __restrict__ pfrag,
             const float* __restrict__ znorm, const float* __restrict__ pnorm,
             float* __restrict__ dists) {
    const int lane = threadIdx.x & 31;
    const int wtile = (blockIdx.x << 3) + (threadIdx.x >> 5);  // 64 tiles
    const int nt2 = wtile & 7, mt2 = wtile >> 3;
    const int col = lane & 15, khalf = lane >> 4, klo = khalf << 3;
    const int brow0 = (mt2 << 5) + col;
    const int brow1 = brow0 + 16;
    v8f c[2][2];
    #pragma unroll
    for (int i = 0; i < 2; ++i)
        #pragma unroll
        for (int j = 0; j < 2; ++j)
            #pragma unroll
            for (int k = 0; k < 8; ++k) c[i][j][k] = 0.f;
    for (int kc = 0; kc < 128; ++kc) {
        const bf16_t* a0 = z + (size_t)brow0 * 4096 + (kc << 5) + klo;
        const bf16_t* a1 = z + (size_t)brow1 * 4096 + (kc << 5) + klo;
        v16bf A0 = cat16(*(const v8bf*)a0, *(const v8bf*)(a0 + 16));
        v16bf A1 = cat16(*(const v8bf*)a1, *(const v8bf*)(a1 + 16));
        const bf16_t* bp = pfrag + (((size_t)kc * 16 + (nt2 << 1)) << 9) + lane * 16;
        v16bf B0 = *(const v16bf*)(bp);
        v16bf B1 = *(const v16bf*)(bp + 512);
        c[0][0] = wmma_bf16(A0, B0, c[0][0]);
        c[0][1] = wmma_bf16(A0, B1, c[0][1]);
        c[1][0] = wmma_bf16(A1, B0, c[1][0]);
        c[1][1] = wmma_bf16(A1, B1, c[1][1]);
    }
    #pragma unroll
    for (int tj = 0; tj < 2; ++tj) {
        const int gp = (nt2 << 5) + (tj << 4) + col;
        const float pn = pnorm[gp];
        #pragma unroll
        for (int ti = 0; ti < 2; ++ti) {
            #pragma unroll
            for (int v = 0; v < 8; ++v) {
                int row = (mt2 << 5) + (ti << 4) + (khalf << 3) + v;
                dists[(size_t)row * 256 + gp] = znorm[row] + pn - 2.f * c[ti][tj][v];
            }
        }
    }
}

// ============== softmin over P=64 per (b,g) -> sm bf16 ==========================
__global__ void k_softmin(const float* __restrict__ dists, bf16_t* __restrict__ sm) {
    int idx = blockIdx.x * 256 + threadIdx.x;    // 1024 rows
    if (idx >= 1024) return;
    int b = idx >> 2, g = idx & 3;
    const float* row = dists + (size_t)b * 256 + g * 64;
    float mn = row[0];
    for (int i = 1; i < 64; ++i) mn = fminf(mn, row[i]);
    float s = 0.f;
    for (int i = 0; i < 64; ++i) s += __expf(mn - row[i]);
    float inv = 1.f / s;
    for (int i = 0; i < 64; ++i) sm[(size_t)idx * 64 + i] = f2bf(__expf(mn - row[i]) * inv);
}

// ============== mixing: mixed(b,d) = 1/4 * sum_g pw(g,d)*[sm_g x proto_g] =======
__global__ __launch_bounds__(256)
void k_mix(const bf16_t* __restrict__ sm, const bf16_t* __restrict__ pmfrag,
           const float* __restrict__ pw, float* __restrict__ mixed,
           bf16_t* __restrict__ mixed_nhwc) {
    const int lane = threadIdx.x & 31;
    const int wtile = (blockIdx.x << 3) + (threadIdx.x >> 5);  // 16*256 tiles
    const int dt = wtile & 255, mt = wtile >> 8;
    const int col = lane & 15, khalf = lane >> 4, klo = khalf << 3;
    const int brow = (mt << 4) + col;
    v8f acc;
    #pragma unroll
    for (int i = 0; i < 8; ++i) acc[i] = 0.f;
    for (int g = 0; g < 4; ++g) {
        v8f cg;
        #pragma unroll
        for (int i = 0; i < 8; ++i) cg[i] = 0.f;
        #pragma unroll
        for (int kc = 0; kc < 2; ++kc) {
            const bf16_t* ap = sm + ((size_t)(brow * 4 + g)) * 64 + (kc << 5) + klo;
            v16bf A = cat16(*(const v8bf*)ap, *(const v8bf*)(ap + 16));
            v16bf B = *(const v16bf*)(pmfrag + (((size_t)((g * 2 + kc) * 256 + dt)) << 9) + lane * 16);
            cg = wmma_bf16(A, B, cg);
        }
        float w = 0.25f * pw[g * 4096 + (dt << 4) + col];
        #pragma unroll
        for (int v = 0; v < 8; ++v) acc[v] += w * cg[v];
    }
    const int d = (dt << 4) + col;
    #pragma unroll
    for (int v = 0; v < 8; ++v) {
        int row = (mt << 4) + (khalf << 3) + v;
        float f = acc[v];
        mixed[(size_t)row * 4096 + d] = f;
        int c = d >> 4, hh = (d >> 2) & 3, ww = d & 3;
        mixed_nhwc[(((size_t)(row * 4 + hh) * 4) + ww) * 256 + c] = f2bf(f);
    }
}

// ============== deconv4: (B,32,32,128) bf16 NHWC -> (B,3,64,64) f32 + sigmoid ===
__global__ void k_deconv4(const bf16_t* __restrict__ in, const float* __restrict__ W,
                          const float* __restrict__ b, float* __restrict__ out) {
    int idx = blockIdx.x * 256 + threadIdx.x;   // B*64*64
    int ow = idx & 63, oh = (idx >> 6) & 63, n = idx >> 12;
    float a0 = b[0], a1 = b[1], a2 = b[2];
    for (int kh = 0; kh < 3; ++kh) {
        int th = oh - 2 + kh;
        if (th & 1) continue;
        int ih = th >> 1;
        if ((unsigned)ih >= 32u) continue;
        for (int kw = 0; kw < 3; ++kw) {
            int tw = ow - 2 + kw;
            if (tw & 1) continue;
            int iw = tw >> 1;
            if ((unsigned)iw >= 32u) continue;
            int r = kh * 3 + kw;
            const bf16_t* xp = in + (((size_t)(n * 32 + ih) * 32 + iw) << 7);
            for (int ci = 0; ci < 128; ++ci) {
                float xv = bf2f(xp[ci]);
                a0 += xv * W[(0 * 128 + ci) * 9 + r];
                a1 += xv * W[(1 * 128 + ci) * 9 + r];
                a2 += xv * W[(2 * 128 + ci) * 9 + r];
            }
        }
    }
    size_t base = (size_t)n * 3 * 4096 + (oh << 6) + ow;
    out[base]          = 1.f / (1.f + __expf(-a0));
    out[base + 4096]   = 1.f / (1.f + __expf(-a1));
    out[base + 8192]   = 1.f / (1.f + __expf(-a2));
}

// =================================================================================
extern "C" void kernel_launch(void* const* d_in, const int* in_sizes, int n_in,
                              void* d_out, int out_size, void* d_ws, size_t ws_size,
                              hipStream_t stream) {
    const float* x   = (const float*)d_in[0];
    const float* We1 = (const float*)d_in[1];  const float* be1 = (const float*)d_in[2];
    const float* We2 = (const float*)d_in[3];  const float* be2 = (const float*)d_in[4];
    const float* We3 = (const float*)d_in[5];  const float* be3 = (const float*)d_in[6];
    const float* We4 = (const float*)d_in[7];  const float* be4 = (const float*)d_in[8];
    const float* Wd1 = (const float*)d_in[9];  const float* bd1 = (const float*)d_in[10];
    const float* Wd2 = (const float*)d_in[11]; const float* bd2 = (const float*)d_in[12];
    const float* Wd3 = (const float*)d_in[13]; const float* bd3 = (const float*)d_in[14];
    const float* Wd4 = (const float*)d_in[15]; const float* bd4 = (const float*)d_in[16];
    const float* proto = (const float*)d_in[17];
    const float* pw    = (const float*)d_in[18];

    float* out = (float*)d_out;
    // output regions (element offsets)
    float* recon_img   = out;                    // 256*3*64*64
    float* recon_proto = out + 3145728;
    float* dists       = out + 6291456;          // 256*256
    float* latent      = out + 6356992;          // 256*4096  (NCHW flat)
    float* proto_out   = out + 7405568;          // 4*64*4096
    float* mixed       = out + 8454144;          // 256*4096

    // workspace carve-up (256B aligned)
    size_t off = 0;
    auto walloc = [&](size_t bytes) -> void* {
        void* p = (void*)((char*)d_ws + off);
        off += (bytes + 255) & ~(size_t)255;
        return p;
    };
    bf16_t* h1   = (bf16_t*)walloc(33554432ull * 2);  // (B,32,32,128)
    bf16_t* h2   = (bf16_t*)walloc( 8388608ull * 2);  // (B,16,16,128)
    bf16_t* h3   = (bf16_t*)walloc( 2097152ull * 2);  // (B,8,8,128)
    bf16_t* latn = (bf16_t*)walloc( 1048576ull * 2);  // (B,4,4,256) NHWC
    bf16_t* zbf  = (bf16_t*)walloc( 1048576ull * 2);  // (B,4096) NCHW-flat
    bf16_t* mixn = (bf16_t*)walloc( 1048576ull * 2);  // (B,4,4,256) NHWC
    bf16_t* sm   = (bf16_t*)walloc(   65536ull * 2);  // (B,4,64)
    bf16_t* t1   = (bf16_t*)walloc( 2097152ull * 2);  // (B,8,8,128)
    bf16_t* t2   = (bf16_t*)walloc( 8388608ull * 2);  // (B,16,16,128)
    bf16_t* t3   = (bf16_t*)walloc(33554432ull * 2);  // (B,32,32,128)
    bf16_t* wf2  = (bf16_t*)walloc(  147456ull * 2);
    bf16_t* wf3  = (bf16_t*)walloc(  147456ull * 2);
    bf16_t* wf4  = (bf16_t*)walloc(  294912ull * 2);
    bf16_t* wfd1 = (bf16_t*)walloc(  294912ull * 2);
    bf16_t* wfd2 = (bf16_t*)walloc(  147456ull * 2);
    bf16_t* wfd3 = (bf16_t*)walloc(  147456ull * 2);
    bf16_t* pdf  = (bf16_t*)walloc( 1048576ull * 2);
    bf16_t* pmf  = (bf16_t*)walloc( 1048576ull * 2);
    float*  znrm = (float*) walloc(256 * 4);
    float*  pnrm = (float*) walloc(256 * 4);
    if (off > ws_size) return;  // insufficient scratch

    // ---- weight / prototype packing ----
    k_pack_w<<<576,  256, 0, stream>>>(We2, wf2, 7, 7, 147456);
    k_pack_w<<<576,  256, 0, stream>>>(We3, wf3, 7, 7, 147456);
    k_pack_w<<<1152, 256, 0, stream>>>(We4, wf4, 7, 8, 294912);
    k_pack_w<<<1152, 256, 0, stream>>>(Wd1, wfd1, 8, 7, 294912);
    k_pack_w<<<576,  256, 0, stream>>>(Wd2, wfd2, 7, 7, 147456);
    k_pack_w<<<576,  256, 0, stream>>>(Wd3, wfd3, 7, 7, 147456);
    k_pack_pd<<<4096, 256, 0, stream>>>(proto, pdf);
    k_pack_pm<<<4096, 256, 0, stream>>>(proto, pmf);

    // ---- encoder ----
    k_conv1<<<131072, 256, 0, stream>>>(x, We1, be1, h1);
    k_convgemm<0,1,false><<<1024, 256, 0, stream>>>(h1, wf2, be2, h2, nullptr, nullptr, 5, 4, 7, 7);
    k_convgemm<0,1,false><<< 256, 256, 0, stream>>>(h2, wf3, be3, h3, nullptr, nullptr, 4, 3, 7, 7);
    k_convgemm<0,2,true ><<< 128, 256, 0, stream>>>(h3, wf4, be4, latn, latent, zbf, 3, 2, 7, 8);

    // ---- prototype distances + softmin + mixing ----
    k_rownorm<<<1, 256, 0, stream>>>(latent, znrm, 4096);
    k_rownorm<<<1, 256, 0, stream>>>(proto,  pnrm, 4096);
    k_dists<<<8, 256, 0, stream>>>(zbf, pdf, znrm, pnrm, dists);
    k_softmin<<<4, 256, 0, stream>>>(dists, sm);
    k_mix<<<512, 256, 0, stream>>>(sm, pmf, pw, mixed, mixn);

    // ---- decoder pass 1: latent -> recon_img ----
    k_convgemm<1,1,false><<< 256, 256, 0, stream>>>(latn, wfd1, bd1, t1, nullptr, nullptr, 2, 3, 8, 7);
    k_convgemm<1,1,false><<<1024, 256, 0, stream>>>(t1,   wfd2, bd2, t2, nullptr, nullptr, 3, 4, 7, 7);
    k_convgemm<1,1,false><<<4096, 256, 0, stream>>>(t2,   wfd3, bd3, t3, nullptr, nullptr, 4, 5, 7, 7);
    k_deconv4<<<4096, 256, 0, stream>>>(t3, Wd4, bd4, recon_img);

    // ---- decoder pass 2: mixed -> recon_proto ----
    k_convgemm<1,1,false><<< 256, 256, 0, stream>>>(mixn, wfd1, bd1, t1, nullptr, nullptr, 2, 3, 8, 7);
    k_convgemm<1,1,false><<<1024, 256, 0, stream>>>(t1,   wfd2, bd2, t2, nullptr, nullptr, 3, 4, 7, 7);
    k_convgemm<1,1,false><<<4096, 256, 0, stream>>>(t2,   wfd3, bd3, t3, nullptr, nullptr, 4, 5, 7, 7);
    k_deconv4<<<4096, 256, 0, stream>>>(t3, Wd4, bd4, recon_proto);

    // ---- prototypes passthrough ----
    hipMemcpyAsync(proto_out, proto, 4ull * 64 * 4096 * sizeof(float),
                   hipMemcpyDeviceToDevice, stream);
}
